// MultiheadAttention_63324997812412
// MI455X (gfx1250) — matrix operand order repaired and checked by reading, not compiled, not run on hardware
//
#include <hip/hip_runtime.h>
#include <hip/hip_bf16.h>

// Problem constants (from reference)
#define T_SEQ 1024
#define BATCH 8
#define EMB   1024
#define HEADS 16
#define HDIM  64
#define MROWS (T_SEQ * BATCH)   // 8192
#define SCALE 0.125f            // HD^-0.5

typedef __attribute__((ext_vector_type(16))) _Float16 v16h;
typedef __attribute__((ext_vector_type(8)))  float    v8f;
typedef __attribute__((ext_vector_type(2)))  _Float16 h2;

__device__ __forceinline__ v8f wmma_f16(v16h a, v16h b, v8f c) {
  return __builtin_amdgcn_wmma_f32_16x16x32_f16(
      /*neg_a=*/false, a, /*neg_b=*/false, b,
      /*c_mod=*/(short)0, c, /*reuse_a=*/false, /*reuse_b=*/false);
}

// A-fragment K offsets, contiguous k order used here (B loads match it):
// lane L: M = L&15, half = L>>4; VGPR g pair at K = 16*half + 2*g
__device__ __forceinline__ void a_koff(int h, int (&koff)[8]) {
#pragma unroll
  for (int g = 0; g < 8; ++g)
    koff[g] = 16 * h + 2 * g;
}

// LDS offset of a __shared__ pointer: flat shared addresses carry the LDS
// offset in the low 32 bits (aperture in the high bits), which is what the
// async-to-LDS instructions consume in their VDST VGPR.
__device__ __forceinline__ unsigned lds_off(const void* p) {
  return (unsigned)(unsigned long long)p;
}

// ---------------------------------------------------------------------------
// Kernel 1: f32 -> f16 conversion; weights stored transposed (N x K) so WMMA
// B-fragments load K-contiguously.
// ---------------------------------------------------------------------------
__global__ __launch_bounds__(256)
void convert_inputs(const float* __restrict__ q,
                    const float* __restrict__ wq, const float* __restrict__ wk,
                    const float* __restrict__ wv, const float* __restrict__ wo,
                    _Float16* __restrict__ X,
                    _Float16* __restrict__ wqt, _Float16* __restrict__ wkt,
                    _Float16* __restrict__ wvt, _Float16* __restrict__ wot) {
  const size_t NQ = (size_t)MROWS * EMB;    // 8M elements
  const size_t NW = (size_t)EMB * EMB;      // 1M elements per weight
  for (size_t i = (size_t)blockIdx.x * blockDim.x + threadIdx.x; i < NQ;
       i += (size_t)gridDim.x * blockDim.x) {
    X[i] = (_Float16)q[i];
    if (i < NW) {
      size_t d = i / EMB, e = i % EMB;      // w[d][e]
      size_t t = e * EMB + d;               // wt[e][d]
      wqt[t] = (_Float16)wq[i];
      wkt[t] = (_Float16)wk[i];
      wvt[t] = (_Float16)wv[i];
      wot[t] = (_Float16)wo[i];
    }
  }
}

// ---------------------------------------------------------------------------
// Kernel 2: WMMA GEMM  C[M=8192,N=1024] = A[M,K=1024] * Bt[N,K]^T
// Block: 256 thr = 8 waves, each wave does a 16x64 tile -> block tile 128x64.
// The 64x32 B tile shared by all 8 waves is staged per K-step into LDS with
// CDNA5 async copies (global_load_async_to_lds_b128, ASYNCcnt), then WMMA
// B-fragments are read as 2x ds_load_b128 per fragment.
// mode 0: f32 row-major out           (final output projection)
// mode 1: f16 head-major (b,h,t,hd)   (Q with scale, K)
// mode 2: f16 head-major T (b,h,hd,t) (V transposed for P@V B-operand)
// ---------------------------------------------------------------------------
__global__ __launch_bounds__(256)
void gemm_wmma(const _Float16* __restrict__ A,
               const _Float16* __restrict__ Bt,
               void* __restrict__ out, int mode, float scale) {
  __shared__ __align__(16) _Float16 Bs[64 * 32];   // 4 KB: Bs[n][k], K-contig
  const int tid  = threadIdx.x;
  const int lane = tid & 31;
  const int wave = tid >> 5;
  const int h    = lane >> 4;
  const int ln   = lane & 15;
  const int tileM = blockIdx.y * 128 + wave * 16;
  const int tileN = blockIdx.x * 64;

  int koff[8];
  a_koff(h, koff);

  // Per-thread slice of the async B-tile stage: 256 threads x 16B = 4 KB.
  const int bn = tid >> 2;          // 0..63  (column within tile)
  const int bk = (tid & 3) * 8;     // 0,8,16,24 (k offset, halves)
  const _Float16* bsrc0 = Bt + (size_t)(tileN + bn) * EMB + bk;
  const unsigned bdst   = lds_off(&Bs[bn * 32 + bk]);

  const _Float16* arow = A + (size_t)(tileM + ln) * EMB;
  v8f acc[4] = {};

  for (int kk = 0; kk < EMB; kk += 32) {
    // ---- async stage B tile (64 x 32 f16) into LDS ----
    {
      unsigned long long ga = (unsigned long long)(bsrc0 + kk);
      asm volatile("global_load_async_to_lds_b128 %0, %1, off"
                   :: "v"(bdst), "v"(ga) : "memory");
    }
    asm volatile("s_wait_asynccnt 0" ::: "memory");
    __syncthreads();

    // ---- A fragment from global (16 contiguous halves in 2 chunks) ----
    v16h a;
#pragma unroll
    for (int g = 0; g < 8; ++g) {
      h2 p = *(const h2*)(arow + kk + koff[g]);
      a[2 * g] = p.x; a[2 * g + 1] = p.y;
    }

    // ---- 4 WMMAs, B fragments from LDS ----
#pragma unroll
    for (int j = 0; j < 4; ++j) {
      const _Float16* bls = Bs + (j * 16 + ln) * 32 + 16 * h;  // 16 halves
      v16h b;
#pragma unroll
      for (int g = 0; g < 8; ++g) {
        h2 p = *(const h2*)(bls + 2 * g);
        b[2 * g] = p.x; b[2 * g + 1] = p.y;
      }
      acc[j] = wmma_f16(a, b, acc[j]);
    }
    __syncthreads();   // protect Bs against next iteration's async writes
  }

  if (mode == 0) {
    float* o = (float*)out;
#pragma unroll
    for (int j = 0; j < 4; ++j)
#pragma unroll
      for (int i = 0; i < 8; ++i) {
        int m = tileM + i + 8 * h;
        int n = tileN + j * 16 + ln;
        o[(size_t)m * EMB + n] = acc[j][i] * scale;
      }
  } else {
    _Float16* o = (_Float16*)out;
#pragma unroll
    for (int j = 0; j < 4; ++j)
#pragma unroll
      for (int i = 0; i < 8; ++i) {
        int m = tileM + i + 8 * h;          // m = t*B + b
        int n = tileN + j * 16 + ln;        // n = hc*64 + hd
        int t = m >> 3, bb = m & 7;
        int hc = n >> 6, hd = n & 63;
        size_t addr = (mode == 1)
            ? ((size_t)((bb * HEADS + hc) * T_SEQ + t) * HDIM + hd)
            : ((size_t)((bb * HEADS + hc) * HDIM + hd) * T_SEQ + t);
        o[addr] = (_Float16)(acc[j][i] * scale);
      }
  }
}

// ---------------------------------------------------------------------------
// Kernel 3: flash attention. grid = (B*H, T/128), 8 waves/block, wave = 16 q
// rows. Score tiles live in WMMA accumulators; bias is preloaded into C.
// P is re-laid-out C->A through a 1KB/wave LDS bounce.
// ---------------------------------------------------------------------------
__global__ __launch_bounds__(256)
void flash_attn(const _Float16* __restrict__ Qh,   // (B*H, T, HD) f16, pre-scaled
                const _Float16* __restrict__ Kh,   // (B*H, T, HD) f16
                const _Float16* __restrict__ Vt,   // (B*H, HD, T) f16
                const float* __restrict__ bias,    // (1, H, T, T) f32
                const unsigned char* __restrict__ mask, // (B, T) bool
                _Float16* __restrict__ attn) {     // (T, B, E) f16
  __shared__ __align__(16) _Float16 pbuf[8 * 16 * 32];
  const int lane = threadIdx.x & 31;
  const int wave = threadIdx.x >> 5;
  const int h    = lane >> 4;
  const int ln   = lane & 15;
  const int bh   = blockIdx.x;
  const int bb   = bh / HEADS, hh = bh % HEADS;
  const int row0 = blockIdx.y * 128 + wave * 16;   // q row (t) tile base

  int koff[8];
  a_koff(h, koff);

  // Resident Q fragments (K = hd: 0..31 and 32..63)
  const _Float16* qrow = Qh + (size_t)(bh * T_SEQ + row0 + ln) * HDIM;
  v16h aq0, aq1;
#pragma unroll
  for (int g = 0; g < 8; ++g) {
    h2 p0 = *(const h2*)(qrow + koff[g]);
    h2 p1 = *(const h2*)(qrow + 32 + koff[g]);
    aq0[2 * g] = p0.x; aq0[2 * g + 1] = p0.y;
    aq1[2 * g] = p1.x; aq1[2 * g + 1] = p1.y;
  }

  float mrow[8], lrow[8];
  v8f o[4] = {};
#pragma unroll
  for (int i = 0; i < 8; ++i) { mrow[i] = -__builtin_inff(); lrow[i] = 0.0f; }

  _Float16* myp = pbuf + wave * (16 * 32);

  for (int s0 = 0; s0 < T_SEQ; s0 += 32) {
    // ---- scores: S = bias + Q K^T, two 16x16 tiles ----
    v8f sc[2];
#pragma unroll
    for (int jt = 0; jt < 2; ++jt) {
      int scol = s0 + jt * 16 + ln;
#pragma unroll
      for (int i = 0; i < 8; ++i) {
        int t = row0 + i + 8 * h;
        sc[jt][i] = bias[(size_t)(hh * T_SEQ + t) * T_SEQ + scol];
      }
      const _Float16* krow = Kh + (size_t)(bh * T_SEQ + scol) * HDIM + 16 * h;
      v16h bk0, bk1;
#pragma unroll
      for (int g = 0; g < 8; ++g) {
        h2 p0 = *(const h2*)(krow + 2 * g);
        h2 p1 = *(const h2*)(krow + 32 + 2 * g);
        bk0[2 * g] = p0.x; bk0[2 * g + 1] = p0.y;
        bk1[2 * g] = p1.x; bk1[2 * g + 1] = p1.y;
      }
      sc[jt] = wmma_f16(aq0, bk0, sc[jt]);
      sc[jt] = wmma_f16(aq1, bk1, sc[jt]);
      if (mask[bb * T_SEQ + scol]) {
#pragma unroll
        for (int i = 0; i < 8; ++i) sc[jt][i] = -__builtin_inff();
      }
    }

    // ---- online softmax (row reductions across the 16-lane half) ----
#pragma unroll
    for (int i = 0; i < 8; ++i) {
      float rm = fmaxf(sc[0][i], sc[1][i]);
#pragma unroll
      for (int d = 1; d < 16; d <<= 1) rm = fmaxf(rm, __shfl_xor(rm, d, 32));
      float newm = fmaxf(mrow[i], rm);
      float alpha, p0, p1;
      if (newm == -__builtin_inff()) {
        alpha = 1.0f; p0 = 0.0f; p1 = 0.0f;
      } else {
        alpha = __expf(mrow[i] - newm);    // exp(-inf)=0 on first live tile
        p0 = (sc[0][i] == -__builtin_inff()) ? 0.0f : __expf(sc[0][i] - newm);
        p1 = (sc[1][i] == -__builtin_inff()) ? 0.0f : __expf(sc[1][i] - newm);
      }
      mrow[i] = newm;
      float ps = p0 + p1;
#pragma unroll
      for (int d = 1; d < 16; d <<= 1) ps += __shfl_xor(ps, d, 32);
      lrow[i] = lrow[i] * alpha + ps;
#pragma unroll
      for (int j = 0; j < 4; ++j) o[j][i] *= alpha;
      // P in (m,k) order to LDS: m = i + 8h, k = jt*16 + ln
      myp[(i + 8 * h) * 32 + ln]      = (_Float16)p0;
      myp[(i + 8 * h) * 32 + 16 + ln] = (_Float16)p1;
    }
    asm volatile("s_wait_dscnt 0" ::: "memory");
    __builtin_amdgcn_wave_barrier();

    // ---- P (16x32) as A fragment ----
    v16h ap;
#pragma unroll
    for (int g = 0; g < 8; ++g) {
      h2 p = *(const h2*)(myp + ln * 32 + koff[g]);
      ap[2 * g] = p.x; ap[2 * g + 1] = p.y;
    }

    // ---- O += P @ V ----
#pragma unroll
    for (int j = 0; j < 4; ++j) {
      const _Float16* vrow =
          Vt + (size_t)(bh * HDIM + j * 16 + ln) * T_SEQ + s0 + 16 * h;
      v16h bv;
#pragma unroll
      for (int g = 0; g < 8; ++g) {
        h2 p = *(const h2*)(vrow + 2 * g);
        bv[2 * g] = p.x; bv[2 * g + 1] = p.y;
      }
      o[j] = wmma_f16(ap, bv, o[j]);
    }
    __builtin_amdgcn_wave_barrier();   // keep LDS read-before-next-write order
  }

  // ---- normalize + store (t, b, h*64+hd) f16 ----
#pragma unroll
  for (int j = 0; j < 4; ++j)
#pragma unroll
    for (int i = 0; i < 8; ++i) {
      int t = row0 + i + 8 * h;
      int e = hh * HDIM + j * 16 + ln;
      float denom = lrow[i];
      float val = (denom > 0.0f) ? (o[j][i] / denom) : 0.0f;
      attn[(size_t)(t * BATCH + bb) * EMB + e] = (_Float16)val;
    }
}

// ---------------------------------------------------------------------------
// Host launcher
// ---------------------------------------------------------------------------
extern "C" void kernel_launch(void* const* d_in, const int* in_sizes, int n_in,
                              void* d_out, int out_size, void* d_ws, size_t ws_size,
                              hipStream_t stream) {
  const float*         q    = (const float*)d_in[0];
  const unsigned char* mask = (const unsigned char*)d_in[1];  // jax bool -> 1B
  const float*         bias = (const float*)d_in[2];
  const float*         wq   = (const float*)d_in[3];
  const float*         wk   = (const float*)d_in[4];
  const float*         wv   = (const float*)d_in[5];
  const float*         wo   = (const float*)d_in[6];

  char* ws = (char*)d_ws;
  const size_t MB = (size_t)1 << 20;
  _Float16* X    = (_Float16*)(ws + 0 * MB);    // 16 MB: query f16 (8192x1024)
  _Float16* wqt  = (_Float16*)(ws + 16 * MB);   //  2 MB each: transposed f16 weights
  _Float16* wkt  = (_Float16*)(ws + 18 * MB);
  _Float16* wvt  = (_Float16*)(ws + 20 * MB);
  _Float16* wot  = (_Float16*)(ws + 22 * MB);
  _Float16* Qh   = (_Float16*)(ws + 24 * MB);   // 16 MB: (B*H,T,HD)
  _Float16* Kh   = (_Float16*)(ws + 40 * MB);   // 16 MB
  _Float16* Vt   = (_Float16*)(ws + 56 * MB);   // 16 MB: (B*H,HD,T)
  _Float16* attn = (_Float16*)(ws + 72 * MB);   // 16 MB: (T,B,E)

  convert_inputs<<<4096, 256, 0, stream>>>(q, wq, wk, wv, wo,
                                           X, wqt, wkt, wvt, wot);

  dim3 ggrid(EMB / 64, MROWS / 128);  // (16, 64)
  gemm_wmma<<<ggrid, 256, 0, stream>>>(X, wqt, (void*)Qh, 1, SCALE);
  gemm_wmma<<<ggrid, 256, 0, stream>>>(X, wkt, (void*)Kh, 1, 1.0f);
  gemm_wmma<<<ggrid, 256, 0, stream>>>(X, wvt, (void*)Vt, 2, 1.0f);

  flash_attn<<<dim3(BATCH * HEADS, T_SEQ / 128), 256, 0, stream>>>(
      Qh, Kh, Vt, bias, mask, attn);

  gemm_wmma<<<ggrid, 256, 0, stream>>>(attn, wot, d_out, 0, 1.0f);
}
